// S4_2619930050875
// MI455X (gfx1250) — compile-verified, matrix operand order checked by reading, not run on hardware
//
#include <hip/hip_runtime.h>
#include <hip/hip_bf16.h>

// ---- problem constants ----
#define HDIM 1024
#define N2   32
#define BSZ  8
#define LSEQ 2048
#define FQ   1025          // L/2+1
#define KPAD 1056          // FQ padded to multiple of 32
#define KCAT (2*KPAD)      // 2112 : [re | im]
#define NTOT (BSZ*LSEQ)    // 16384
#define PI_F 3.14159265358979f

typedef _Float16 h16 __attribute__((ext_vector_type(16)));
typedef _Float16 h8  __attribute__((ext_vector_type(8)));
typedef float    f8  __attribute__((ext_vector_type(8)));

// vector-of-4-int types matching the async builtin's parameter types
typedef int i4v __attribute__((vector_size(4 * sizeof(int))));
typedef __attribute__((address_space(1))) i4v gi4;   // global
typedef __attribute__((address_space(3))) i4v li4;   // LDS

union FragH { h16 v; h8 h2[2]; _Float16 e[16]; };

__device__ __forceinline__ f8 wmma_f16(h16 a, h16 b, f8 c) {
    // D = A x B + C, 16x16x32 f16 -> f32, wave32
    return __builtin_amdgcn_wmma_f32_16x16x32_f16(
        /*neg_a*/false, a, /*neg_b*/false, b,
        /*c_mod*/(short)0, c, /*reuse_a*/false, /*reuse_b*/false);
}

// ---- CDNA5 async global->LDS copy (ASYNCcnt path), guarded fallback ----
__device__ __forceinline__ void async_copy16(const void* g, void* l) {
#if defined(__has_builtin) && __has_builtin(__builtin_amdgcn_global_load_async_to_lds_b128)
    __builtin_amdgcn_global_load_async_to_lds_b128((gi4*)g, (li4*)l, 0, 0);
#else
    *(h8*)l = *(const h8*)g;
#endif
}
__device__ __forceinline__ void async_wait0() {
#if defined(__has_builtin) && __has_builtin(__builtin_amdgcn_s_wait_asynccnt)
    __builtin_amdgcn_s_wait_asynccnt(0);
#endif
}

// Load a 16x32 f16 tile (row-major, ld=32) from LDS into an A/B fragment.
// ISA layout: lane&15 = row; lane>>4 selects K-chunks {0..7,16..23} vs {8..15,24..31}.
__device__ __forceinline__ h16 load_frag(const _Float16* tile, int lane) {
    int row = lane & 15;
    int hl  = lane >> 4;
    const _Float16* p = tile + row * 32 + hl * 8;
    FragH f;
    f.h2[0] = *(const h8*)(p);
    f.h2[1] = *(const h8*)(p + 16);
    return f.v;
}

// ------------------------------------------------------------------
// K1: Cauchy kernel + rank-1 Woodbury -> kf, written as f16 into the
//     concatenated GEMM A-matrix  Abuf[h][0..1055]=re, [1056..2111]=im
// ------------------------------------------------------------------
__global__ void s4_cauchy_woodbury(
    const float* __restrict__ log_dt,
    const float* __restrict__ Lre, const float* __restrict__ Lim,
    const float* __restrict__ Pre, const float* __restrict__ Pim,
    const float* __restrict__ Bre, const float* __restrict__ Bim,
    const float* __restrict__ Cre, const float* __restrict__ Cim,
    _Float16* __restrict__ Abuf)
{
    int h = blockIdx.x;
    int tid = threadIdx.x;                 // 128 threads
    __shared__ float slr[N2], sli[N2], spr[N2], spi[N2],
                     sbr[N2], sbi[N2], scr[N2], sci[N2];
    if (tid < N2) {
        slr[tid] = Lre[h*N2+tid]; sli[tid] = Lim[h*N2+tid];
        spr[tid] = Pre[h*N2+tid]; spi[tid] = Pim[h*N2+tid];
        sbr[tid] = Bre[h*N2+tid]; sbi[tid] = Bim[h*N2+tid];
        scr[tid] = Cre[h*N2+tid]; sci[tid] = Cim[h*N2+tid];
    }
    __syncthreads();
    float dt = __expf(log_dt[h]);

    for (int f = tid; f < KPAD; f += 128) {
        float outre = 0.f, outim = 0.f;
        if (f < 1024) {                    // f==1024 (Nyquist) & padding -> 0
            float th = -2.f * PI_F * (float)f / (float)LSEQ;
            float owr, owi; __sincosf(th, &owi, &owr);
            // z = 2(1-omega)/(1+omega)
            float dr = 1.f + owr, di = owi;
            float dd = dr*dr + di*di;
            float nr = 1.f - owr, ni = -owi;
            float zr = 2.f*(nr*dr + ni*di)/dd;
            float zi = 2.f*(ni*dr - nr*di)/dd;

            float r00r=0,r00i=0, r01r=0,r01i=0, r10r=0,r10i=0, r11r=0,r11i=0;
            #pragma unroll 4
            for (int n = 0; n < N2; ++n) {
                float wr = slr[n]*dt, wi = sli[n]*dt;
                float ar = zr - wr,  ai = zi - wi;
                float ia = 1.f/(ar*ar + ai*ai);
                float q1r =  ar*ia, q1i = -ai*ia;          // 1/(z-w)
                float br2 = zr - wr, bi2 = zi + wi;
                float ib = 1.f/(br2*br2 + bi2*bi2);
                float q2r =  br2*ib, q2i = -bi2*ib;        // 1/(z-conj(w))

                float cr0 = scr[n], ci0 = sci[n];
                float brr = sbr[n], bii = sbi[n];
                float prr = spr[n], pii = spi[n];
                float v00r = cr0*brr - ci0*bii, v00i = cr0*bii + ci0*brr; // C*B
                float v01r = cr0*prr - ci0*pii, v01i = cr0*pii + ci0*prr; // C*P
                float v10r = prr*brr + pii*bii, v10i = prr*bii - pii*brr; // conj(P)*B
                float v11r = prr*prr + pii*pii, v11i = 0.f;               // conj(P)*P

                #define CACC(rr,ri,vr,vi)                                   \
                    rr += (vr)*q1r - (vi)*q1i + (vr)*q2r + (vi)*q2i;        \
                    ri += (vr)*q1i + (vi)*q1r + (vr)*q2i - (vi)*q2r;
                CACC(r00r,r00i,v00r,v00i)
                CACC(r01r,r01i,v01r,v01i)
                CACC(r10r,r10i,v10r,v10i)
                CACC(r11r,r11i,v11r,v11i)
                #undef CACC
            }
            r00r*=dt; r00i*=dt; r01r*=dt; r01i*=dt;
            r10r*=dt; r10i*=dt; r11r*=dt; r11i*=dt;
            // kf = r00 - r01*r10/(1+r11)
            float w1r = 1.f + r11r, w1i = r11i;
            float wd  = w1r*w1r + w1i*w1i;
            float numr = r01r*r10r - r01i*r10i;
            float numi = r01r*r10i + r01i*r10r;
            float cr2 = (numr*w1r + numi*w1i)/wd;
            float ci2 = (numi*w1r - numr*w1i)/wd;
            float kfr = r00r - cr2, kfi = r00i - ci2;
            // *= 2/(1+omega)
            float sc = 2.f/dd;
            outre = (kfr*dr + kfi*di)*sc;
            outim = (kfi*dr - kfr*di)*sc;
        }
        Abuf[(size_t)h*KCAT + f]        = (_Float16)outre;
        Abuf[(size_t)h*KCAT + KPAD + f] = (_Float16)outim;
    }
}

// ------------------------------------------------------------------
// K2: iDFT basis (2112 x 2048) f16 : rows [0,1056)=w*cos/n, [1056,2112)=-w*sin/n
// ------------------------------------------------------------------
__global__ void s4_gen_basis(_Float16* __restrict__ Bm)
{
    size_t idx = (size_t)blockIdx.x * 256 + threadIdx.x;
    if (idx >= (size_t)KCAT * LSEQ) return;
    int row = (int)(idx >> 11);
    int l   = (int)(idx & 2047);
    int f   = (row < KPAD) ? row : row - KPAD;
    float val = 0.f;
    if (f < FQ) {
        float w = (f == 0 || f == 1024) ? 1.f : 2.f;
        float th = 2.f * PI_F * (float)f * (float)l / (float)LSEQ;
        float s, c; __sincosf(th, &s, &c);
        val = ((row < KPAD) ? w*c : -w*s) * (1.f/(float)LSEQ);
    }
    Bm[idx] = (_Float16)val;
}

// ------------------------------------------------------------------
// K3: f32 -> f16 convert (u and W_out)
// ------------------------------------------------------------------
__global__ void s4_cvt_f16(const float* __restrict__ src, _Float16* __restrict__ dst, size_t n)
{
    size_t i = (size_t)blockIdx.x * 256 + threadIdx.x;
    if (i < n) dst[i] = (_Float16)src[i];
}

// ------------------------------------------------------------------
// K4: iDFT GEMM:  k16 (1024 x 2048) = Abuf (1024 x 2112) @ basis (2112 x 2048)
//     workgroup = 128 thr (4 waves), 64x64 tile, K chunk 32
//     A staged via async global->LDS; B read coalesced, stored transposed
// ------------------------------------------------------------------
__global__ void s4_idft_gemm(const _Float16* __restrict__ A,
                             const _Float16* __restrict__ Bm,
                             _Float16* __restrict__ K16)
{
    __shared__ __align__(16) _Float16 As[64*32];
    __shared__ __align__(16) _Float16 Bs[64*32];   // transposed: Bs[n][k]
    int m0 = blockIdx.y * 64;
    int n0 = blockIdx.x * 64;
    int tid = threadIdx.x;
    int wave = tid >> 5, lane = tid & 31;
    f8 acc[4] = {};

    int alr = lane >> 2;            // 0..7 : row within 8-row segment
    int alc = (lane & 3) * 8;       // 16B chunk within 32-half row
    int bk  = tid >> 2;             // 0..31 : K row for B staging
    int bn  = (tid & 3) * 16;       // n offset for B staging

    for (int k0 = 0; k0 < KCAT; k0 += 32) {
        __syncthreads();
        // --- A tile 64x32: 8 x 512B wave-copies via ASYNCcnt path ---
        #pragma unroll
        for (int i = 0; i < 2; ++i) {
            int r = wave * 16 + i * 8 + alr;
            async_copy16(A + (size_t)(m0 + r) * KCAT + k0 + alc,
                         As + r * 32 + alc);
        }
        // --- B tile 32x64 -> transposed Bs[n][k], coalesced reads ---
        {
            const _Float16* src = Bm + (size_t)(k0 + bk) * LSEQ + n0 + bn;
            h8 v0 = *(const h8*)(src);
            h8 v1 = *(const h8*)(src + 8);
            #pragma unroll
            for (int j = 0; j < 8; ++j) {
                Bs[(bn + j) * 32 + bk]     = v0[j];
                Bs[(bn + 8 + j) * 32 + bk] = v1[j];
            }
        }
        async_wait0();
        __syncthreads();
        h16 af = load_frag(As + wave * 16 * 32, lane);
        #pragma unroll
        for (int nt = 0; nt < 4; ++nt) {
            h16 bf = load_frag(Bs + nt * 16 * 32, lane);
            acc[nt] = wmma_f16(af, bf, acc[nt]);
        }
    }
    int hl = lane >> 4;
    int cl = lane & 15;
    #pragma unroll
    for (int nt = 0; nt < 4; ++nt)
        #pragma unroll
        for (int v = 0; v < 8; ++v) {
            int r = m0 + wave*16 + v + 8*hl;
            K16[(size_t)r * LSEQ + n0 + nt*16 + cl] = (_Float16)acc[nt][v];
        }
}

// ------------------------------------------------------------------
// K5: causal conv via block-Toeplitz WMMA + skip + exact GELU
//     Y[h][b*L+l] = gelu( sum_m k[h,m] u[b,h,l-m] + D[h]*u[b,h,l] )
//     workgroup 256 thr (8 waves) per (h, 128-l block); wave = one 16-row M tile
// ------------------------------------------------------------------
__global__ void s4_conv_gelu(const _Float16* __restrict__ K16,   // 1024 x 2048
                             const _Float16* __restrict__ U16,   // (8,1024,2048)
                             const float*    __restrict__ Uf,    // (8,1024,2048)
                             const float*    __restrict__ Dskip, // (1,1024)
                             _Float16* __restrict__ Y)           // 1024 x 16384
{
    __shared__ __align__(16) _Float16 ksh[64 + LSEQ];  // 64-elem zero guard (causal)
    int h    = blockIdx.y;
    int lblk = blockIdx.x;
    int tid  = threadIdx.x;
    int wave = tid >> 5, lane = tid & 31;

    if (tid < 64) ksh[tid] = (_Float16)0.f;
    // 4KB contiguous stage of k_h via async path: 256 lanes x 16B
    async_copy16(K16 + (size_t)h * LSEQ + tid * 8, ksh + 64 + tid * 8);
    async_wait0();
    __syncthreads();

    int l0 = lblk * 128 + wave * 16;
    int b  = lane & 15;
    int hl = lane >> 4;
    int row = lane & 15;
    const _Float16* ubase = U16 + ((size_t)(b & 7) * HDIM + h) * LSEQ;

    f8 acc = {};
    int cmax = (l0 + 15) >> 5;                 // causal: K chunks with 32c <= l0+15
    for (int c = 0; c <= cmax; ++c) {
        int m0 = c * 32;
        // A fragment: Toeplitz slice A[i][kk] = k[l0+i - m0 - kk] (guard handles <0)
        FragH af;
        int base = 64 + l0 + row - m0;
        #pragma unroll
        for (int j = 0; j < 8; ++j) {
            af.e[j]     = ksh[base - (hl*8 + j)];
            af.e[j + 8] = ksh[base - (16 + hl*8 + j)];
        }
        // B fragment: u columns (batch); lanes with b>=8 are zero padding
        FragH bf;
        if (b < 8) {
            bf.h2[0] = *(const h8*)(ubase + m0 + hl*8);
            bf.h2[1] = *(const h8*)(ubase + m0 + 16 + hl*8);
        } else {
            #pragma unroll
            for (int j = 0; j < 16; ++j) bf.e[j] = (_Float16)0.f;
        }
        acc = wmma_f16(af.v, bf.v, acc);       // EXEC all-ones: no divergence here
    }

    if (b < 8) {
        float dh = Dskip[h];
        const float* uf = Uf + ((size_t)b * HDIM + h) * LSEQ;
        #pragma unroll
        for (int v = 0; v < 8; ++v) {
            int l = l0 + v + 8*hl;
            float y = acc[v] + dh * uf[l];
            y = 0.5f * y * (1.f + erff(y * 0.70710678118f));   // exact GELU
            Y[(size_t)h * NTOT + b * LSEQ + l] = (_Float16)y;
        }
    }
}

// ------------------------------------------------------------------
// K6: output projection + bias + GLU
//     z = W16 (2048x1024) @ Y (1024x16384);  out = z[:1024]*sigmoid(z[1024:])
//     workgroup 256 thr (8 waves): paired 32-row M tiles (o0, o0+1024), N tile 128
//     4 WMMAs per wave per K chunk; A via async LDS fill; B coalesced+transposed
// ------------------------------------------------------------------
__global__ void s4_proj_glu(const _Float16* __restrict__ W16,   // 2048 x 1024
                            const _Float16* __restrict__ Y,     // 1024 x 16384
                            const float*    __restrict__ bias,  // 2048
                            float* __restrict__ Out)            // (8,1024,2048)
{
    __shared__ __align__(16) _Float16 Aa[32*32];
    __shared__ __align__(16) _Float16 Ag[32*32];
    __shared__ __align__(16) _Float16 Bs[128*32];  // transposed: Bs[n][k]
    int o0 = blockIdx.y * 32;
    int n0 = blockIdx.x * 128;
    int tid = threadIdx.x;
    int wave = tid >> 5, lane = tid & 31;
    f8 acc_a0 = {}, acc_a1 = {}, acc_g0 = {}, acc_g1 = {};

    // A staging: waves 0-3 fill Aa, waves 4-7 fill Ag (one 512B async op each)
    int arow = (wave & 3) * 8 + (lane >> 2);        // 0..31
    int acol = (lane & 3) * 8;                      // 16B chunk in 32-half row
    const _Float16* asrc =
        W16 + (size_t)(o0 + ((wave < 4) ? 0 : 1024) + arow) * HDIM + acol;
    _Float16* adst = ((wave < 4) ? Aa : Ag) + arow * 32 + acol;
    // B staging: coalesced read of 32x128 tile, transposed scatter into LDS
    int bk = tid >> 3;                              // 0..31
    int bn = (tid & 7) * 16;                        // 0..112

    for (int k0 = 0; k0 < HDIM; k0 += 32) {
        __syncthreads();
        async_copy16(asrc + k0, adst);
        {
            const _Float16* src = Y + (size_t)(k0 + bk) * NTOT + n0 + bn;
            h8 v0 = *(const h8*)(src);
            h8 v1 = *(const h8*)(src + 8);
            #pragma unroll
            for (int j = 0; j < 8; ++j) {
                Bs[(bn + j) * 32 + bk]     = v0[j];
                Bs[(bn + 8 + j) * 32 + bk] = v1[j];
            }
        }
        if (k0 + 32 < HDIM)
            __builtin_prefetch(&Y[(size_t)(k0 + 32 + bk) * NTOT + n0 + bn], 0, 3);
        async_wait0();
        __syncthreads();
        h16 bf = load_frag(Bs + wave * 16 * 32, lane);
        h16 a0 = load_frag(Aa,           lane);
        h16 a1 = load_frag(Aa + 16 * 32, lane);
        h16 g0 = load_frag(Ag,           lane);
        h16 g1 = load_frag(Ag + 16 * 32, lane);
        acc_a0 = wmma_f16(a0, bf, acc_a0);
        acc_a1 = wmma_f16(a1, bf, acc_a1);
        acc_g0 = wmma_f16(g0, bf, acc_g0);
        acc_g1 = wmma_f16(g1, bf, acc_g1);
    }

    int hl  = lane >> 4;
    int col = n0 + wave * 16 + (lane & 15);
    int bidx = col >> 11;                       // /2048
    int l    = col & 2047;
    float* obase = Out + (size_t)bidx * HDIM * LSEQ + l;
    #pragma unroll
    for (int v = 0; v < 8; ++v) {
        int oa0 = o0 + v + 8*hl;
        int oa1 = o0 + 16 + v + 8*hl;
        float za0 = acc_a0[v] + bias[oa0];
        float zg0 = acc_g0[v] + bias[oa0 + 1024];
        float za1 = acc_a1[v] + bias[oa1];
        float zg1 = acc_g1[v] + bias[oa1 + 1024];
        obase[(size_t)oa0 * LSEQ] = za0 * (1.f / (1.f + __expf(-zg0)));
        obase[(size_t)oa1 * LSEQ] = za1 * (1.f / (1.f + __expf(-zg1)));
    }
}

// ------------------------------------------------------------------
extern "C" void kernel_launch(void* const* d_in, const int* in_sizes, int n_in,
                              void* d_out, int out_size, void* d_ws, size_t ws_size,
                              hipStream_t stream)
{
    const float* u      = (const float*)d_in[0];
    const float* log_dt = (const float*)d_in[1];
    const float* Lre    = (const float*)d_in[2];
    const float* Lim    = (const float*)d_in[3];
    const float* Pre    = (const float*)d_in[4];
    const float* Pim    = (const float*)d_in[5];
    const float* Bre    = (const float*)d_in[6];
    const float* Bim    = (const float*)d_in[7];
    const float* Cre    = (const float*)d_in[8];
    const float* Cim    = (const float*)d_in[9];
    const float* Dsk    = (const float*)d_in[10];
    const float* Wout   = (const float*)d_in[11];
    const float* bout   = (const float*)d_in[12];
    float* out = (float*)d_out;

    // workspace carve-up (all 256B aligned)
    char* ws = (char*)d_ws;
    size_t off = 0;
    auto take = [&](size_t bytes) { char* p = ws + off; off = (off + bytes + 255) & ~(size_t)255; return p; };
    _Float16* Abuf  = (_Float16*)take((size_t)HDIM * KCAT * 2);       // 4.3 MB
    _Float16* basis = (_Float16*)take((size_t)KCAT * LSEQ * 2);       // 8.7 MB
    _Float16* k16   = (_Float16*)take((size_t)HDIM * LSEQ * 2);       // 4.2 MB
    _Float16* u16   = (_Float16*)take((size_t)BSZ * HDIM * LSEQ * 2); // 33.6 MB
    _Float16* w16   = (_Float16*)take((size_t)2 * HDIM * HDIM * 2);   // 4.2 MB
    _Float16* yact  = (_Float16*)take((size_t)HDIM * NTOT * 2);       // 33.6 MB
    (void)ws_size;

    // 1) Cauchy + Woodbury -> Abuf
    s4_cauchy_woodbury<<<HDIM, 128, 0, stream>>>(log_dt, Lre, Lim, Pre, Pim,
                                                 Bre, Bim, Cre, Cim, Abuf);
    // 2) iDFT basis
    {
        size_t n = (size_t)KCAT * LSEQ;
        s4_gen_basis<<<(unsigned)((n + 255) / 256), 256, 0, stream>>>(basis);
    }
    // 3) converts
    {
        size_t nu = (size_t)BSZ * HDIM * LSEQ;
        s4_cvt_f16<<<(unsigned)((nu + 255) / 256), 256, 0, stream>>>(u, u16, nu);
        size_t nw = (size_t)2 * HDIM * HDIM;
        s4_cvt_f16<<<(unsigned)((nw + 255) / 256), 256, 0, stream>>>(Wout, w16, nw);
    }
    // 4) k = irfft(kf) as WMMA GEMM
    s4_idft_gemm<<<dim3(LSEQ / 64, HDIM / 64), 128, 0, stream>>>(Abuf, basis, k16);
    // 5) causal conv (block-Toeplitz WMMA) + skip + GELU
    s4_conv_gelu<<<dim3(LSEQ / 128, HDIM), 256, 0, stream>>>(k16, u16, u, Dsk, yact);
    // 6) projection GEMM + bias + GLU (paired tiles, fused)
    s4_proj_glu<<<dim3(NTOT / 128, HDIM / 32), 256, 0, stream>>>(w16, yact, bout, out);
    (void)out_size; (void)n_in; (void)in_sizes;
}
